// MultiHeadAttentionBlock_71708773974538
// MI455X (gfx1250) — matrix operand order repaired and checked by reading, not compiled, not run on hardware
//
#include <hip/hip_runtime.h>

// ---------------------------------------------------------------------------
// MHA block for MI455X (gfx1250, wave32). All matmuls in bf16 WMMA
// (v_wmma_f32_16x16x32_bf16), fp32 accumulation. Whole working set is
// L2-resident (~160MB < 192MB), so the multi-kernel pipeline pays L2, not HBM.
// GEMMs stage tiles in LDS via async global->LDS copies (ASYNCcnt pipeline)
// when the toolchain exposes the gfx1250 builtin; otherwise they stream
// fragments straight from L2.
// ---------------------------------------------------------------------------

typedef __attribute__((ext_vector_type(16))) __bf16 v16bf;
typedef __attribute__((ext_vector_type(8)))  __bf16 v8bf;
typedef __attribute__((ext_vector_type(4)))  __bf16 v4bf;
typedef __attribute__((ext_vector_type(8)))  float  v8f;
typedef __attribute__((ext_vector_type(4)))  int    v4i;

// async global->LDS builtin takes (AS1 v4i*, AS3 v4i*, imm offset, imm cpol)
typedef __attribute__((address_space(1))) v4i as1_v4i;
typedef __attribute__((address_space(3))) v4i as3_v4i;

#define D_MODEL   1024
#define NUM_HEADS 16
#define D_K       64
#define SEQ       2048
#define BATCH     4

#if __has_builtin(__builtin_amdgcn_global_load_async_to_lds_b128)
#define HAVE_ASYNC_LDS 1
#else
#define HAVE_ASYNC_LDS 0
#endif

template<int N>
static __device__ __forceinline__ void wait_async_le() {
#if __has_builtin(__builtin_amdgcn_s_wait_asynccnt)
  __builtin_amdgcn_s_wait_asynccnt(N);
#else
  if (N == 0) asm volatile("s_wait_asynccnt 0x0" ::: "memory");
  else        asm volatile("s_wait_asynccnt 0x4" ::: "memory");
#endif
}

static __device__ inline __bf16 f2bf(float f) {
  union { float f; unsigned u; } a; a.f = f;
  unsigned r = a.u + 0x7FFFu + ((a.u >> 16) & 1u);   // round-to-nearest-even
  union { unsigned short s; __bf16 b; } o; o.s = (unsigned short)(r >> 16);
  return o.b;
}

static __device__ inline v16bf join16(v8bf lo, v8bf hi) {
  union { v16bf v; v8bf h[2]; } u;
  u.h[0] = lo; u.h[1] = hi;
  return u.v;
}

static __device__ inline v8f wmma_bf16(v16bf a, v16bf b, v8f c) {
  return __builtin_amdgcn_wmma_f32_16x16x32_bf16(false, a, false, b,
                                                 (short)0, c, false, false);
}

// ---------------------------------------------------------------------------
// fp32 -> bf16 conversion, 4 elements / thread (sizes are multiples of 1024)
// ---------------------------------------------------------------------------
__global__ __launch_bounds__(256)
void cvt_f32_bf16_kernel(const float* __restrict__ src, __bf16* __restrict__ dst, int n) {
  int i = 4 * (blockIdx.x * blockDim.x + threadIdx.x);
  if (i >= n) return;
  float4 f = *reinterpret_cast<const float4*>(src + i);
  v4bf o;
  o[0] = f2bf(f.x); o[1] = f2bf(f.y); o[2] = f2bf(f.z); o[3] = f2bf(f.w);
  *reinterpret_cast<v4bf*>(dst + i) = o;
}

// ---------------------------------------------------------------------------
// Y[m][n] = sum_k A[m][k] * W[n][k]   (A: Mx1024 bf16 row-major,
//                                      W: 1024x1024 bf16 row-major [n][k])
// Block 128x128 (256 thr = 8 waves, 2x4 wave grid), wave tile 64x32,
// K-step 32, 8 wmma / wave / K-step.
// Async path: double-buffered LDS staging of the 128x32 A and B tiles via
// global_load_async_to_lds_b128 (4 async ops / wave / tile; issue next tile,
// s_wait_asynccnt 4, barrier, compute). Removes the 4x (A) / 2x (B)
// duplicate fragment loads across waves and overlaps copy with WMMA.
// MODE 0: bf16 out, (B,H,S,Dk)     -- q, k
// MODE 1: bf16 out, (B,H,Dk,S)     -- v transposed (for P@V B-fragments)
// MODE 2: f32  out, row-major      -- final projection
// ---------------------------------------------------------------------------
template<int MODE>
__global__ __launch_bounds__(256)
void gemm_bf16_wmma(const __bf16* __restrict__ A, const __bf16* __restrict__ W,
                    void* __restrict__ out) {
  const int tid  = threadIdx.x;
  const int lane = tid & 31;
  const int wave = tid >> 5;
  const int hb   = lane >> 4;          // half of wave (A/B/C layout split)
  const int l16  = lane & 15;
  const int wm   = wave & 1;           // 2 waves along M
  const int wn   = wave >> 1;          // 4 waves along N
  const int row0 = blockIdx.y * 128 + wm * 64;
  const int col0 = blockIdx.x * 128 + wn * 32;

  v8f acc[4][2];
#pragma unroll
  for (int mt = 0; mt < 4; ++mt)
#pragma unroll
    for (int nt = 0; nt < 2; ++nt)
#pragma unroll
      for (int e = 0; e < 8; ++e) acc[mt][nt][e] = 0.f;

#if HAVE_ASYNC_LDS
  // ---------------- async double-buffered LDS pipeline ----------------
  __shared__ __align__(16) __bf16 ldsA[2][128 * 32];   // 8KB per buffer
  __shared__ __align__(16) __bf16 ldsB[2][128 * 32];
  const int blkRow0 = blockIdx.y * 128;
  const int blkCol0 = blockIdx.x * 128;

  auto issue_tile = [&](int t, int buf) {
    const int k0 = t * 32;
#pragma unroll
    for (int j = 0; j < 2; ++j) {                       // 2x 16B chunks/thread
      const int c   = tid + j * 256;
      const int row = c >> 2;                           // 4 chunks per row
      const int ce  = (c & 3) * 8;                      // element offset
      __builtin_amdgcn_global_load_async_to_lds_b128(
          (as1_v4i*)(A + (size_t)(blkRow0 + row) * D_MODEL + k0 + ce),
          (as3_v4i*)(&ldsA[buf][row * 32 + ce]), 0, 0);
      __builtin_amdgcn_global_load_async_to_lds_b128(
          (as1_v4i*)(W + (size_t)(blkCol0 + row) * D_MODEL + k0 + ce),
          (as3_v4i*)(&ldsB[buf][row * 32 + ce]), 0, 0);
    }
  };

  int buf = 0;
  issue_tile(0, 0);
  const int NT = D_MODEL / 32;                          // 32 K-tiles
  for (int t = 0; t < NT; ++t) {
    if (t + 1 < NT) { issue_tile(t + 1, buf ^ 1); wait_async_le<4>(); }
    else            { wait_async_le<0>(); }
    __syncthreads();                                    // all waves' tile done

    v16bf afrag[4], bfrag[2];
#pragma unroll
    for (int mt = 0; mt < 4; ++mt) {
      const __bf16* p = &ldsA[buf][(wm * 64 + mt * 16 + l16) * 32 + hb * 16];
      afrag[mt] = *reinterpret_cast<const v16bf*>(p);
    }
#pragma unroll
    for (int nt = 0; nt < 2; ++nt) {
      const __bf16* p = &ldsB[buf][(wn * 32 + nt * 16 + l16) * 32 + hb * 16];
      bfrag[nt] = *reinterpret_cast<const v16bf*>(p);
    }
#pragma unroll
    for (int mt = 0; mt < 4; ++mt)
#pragma unroll
      for (int nt = 0; nt < 2; ++nt)
        acc[mt][nt] = wmma_bf16(afrag[mt], bfrag[nt], acc[mt][nt]);

    __syncthreads();                                    // done reading buf
    buf ^= 1;
  }
#else
  // ---------------- direct-from-L2 fallback ----------------
  const __bf16* aRow[4];
#pragma unroll
  for (int mt = 0; mt < 4; ++mt)
    aRow[mt] = A + (size_t)(row0 + mt * 16 + l16) * D_MODEL;
  const __bf16* bRow[2];
#pragma unroll
  for (int nt = 0; nt < 2; ++nt)
    bRow[nt] = W + (size_t)(col0 + nt * 16 + l16) * D_MODEL;

  for (int k0 = 0; k0 < D_MODEL; k0 += 32) {
    if (k0 + 256 < D_MODEL)            // global_prefetch_b8 path
      __builtin_prefetch(aRow[0] + k0 + 256, 0, 1);

    v16bf afrag[4], bfrag[2];
#pragma unroll
    for (int mt = 0; mt < 4; ++mt) {
      const __bf16* p = aRow[mt] + k0 + hb * 16;
      afrag[mt] = *reinterpret_cast<const v16bf*>(p);
    }
#pragma unroll
    for (int nt = 0; nt < 2; ++nt)
      bfrag[nt] = *reinterpret_cast<const v16bf*>(bRow[nt] + k0 + hb * 16);
#pragma unroll
    for (int mt = 0; mt < 4; ++mt)
#pragma unroll
      for (int nt = 0; nt < 2; ++nt)
        acc[mt][nt] = wmma_bf16(afrag[mt], bfrag[nt], acc[mt][nt]);
  }
#endif

  // epilogue: C/D layout row = r + 8*hb, col = l16
#pragma unroll
  for (int mt = 0; mt < 4; ++mt) {
#pragma unroll
    for (int nt = 0; nt < 2; ++nt) {
#pragma unroll
      for (int r = 0; r < 8; ++r) {
        const int row = row0 + mt * 16 + r + hb * 8;   // m index (b*S+s)
        const int col = col0 + nt * 16 + l16;          // n index (h*Dk+d)
        const float v = acc[mt][nt][r];
        if (MODE == 2) {
          reinterpret_cast<float*>(out)[(size_t)row * D_MODEL + col] = v;
        } else {
          const int bb = row >> 11, s = row & (SEQ - 1);
          const int hh = col >> 6,  d = col & 63;
          const __bf16 bv = f2bf(v);
          if (MODE == 0)
            reinterpret_cast<__bf16*>(out)[(((size_t)bb * NUM_HEADS + hh) * SEQ + s) * D_K + d] = bv;
          else
            reinterpret_cast<__bf16*>(out)[(((size_t)bb * NUM_HEADS + hh) * D_K + d) * SEQ + s] = bv;
        }
      }
    }
  }
}

// ---------------------------------------------------------------------------
// One 64-key block of causal flash attention for one wave (16 q rows).
// MASKED=false is the steady state: zero compare/cndmask VALU.
// Row max:   __shfl_xor within the 16-lane halves (matches C-tile row split).
// Row sum:   WMMA against an all-ones B matrix -- P*1 lands each row's sum
//            in the C tile in exactly the lrow[] register layout.
// ---------------------------------------------------------------------------
template<bool MASKED>
static __device__ __forceinline__
void attn_kblock(int kb, int qrow0, int l16, int hb,
                 const __bf16* __restrict__ kbase,
                 const __bf16* __restrict__ vbase,
                 __bf16* __restrict__ pw,
                 const v16bf (&qa)[2], const v16bf& ones,
                 v8f (&acc)[4], float (&mrow)[8], float (&lrow)[8]) {
  const float scale = 0.125f;           // 1/sqrt(D_K)

  // ---- scores: S = Q . K^T, 16x64 as four 16x16 C tiles (8 wmma)
  v8f sc[4];
#pragma unroll
  for (int nt = 0; nt < 4; ++nt) {
    v8f c;
#pragma unroll
    for (int e = 0; e < 8; ++e) c[e] = 0.f;
    const __bf16* kp = kbase + (size_t)(kb + nt * 16 + l16) * D_K + hb * 16;
    v16bf b0 = *reinterpret_cast<const v16bf*>(kp);        // dims 0..31
    v16bf b1 = *reinterpret_cast<const v16bf*>(kp + 32);   // dims 32..63
    c = wmma_bf16(qa[0], b0, c);
    c = wmma_bf16(qa[1], b1, c);
    sc[nt] = c;
  }

  // ---- streaming softmax stats (row = r + 8*hb, col = nt*16 + l16)
  float alpha[8];
#pragma unroll
  for (int r = 0; r < 8; ++r) {
    float s[4];
#pragma unroll
    for (int nt = 0; nt < 4; ++nt) {
      s[nt] = sc[nt][r] * scale;
      if (MASKED) {
        const int qg = qrow0 + r + hb * 8;
        if (kb + nt * 16 + l16 > qg) s[nt] = -1e9f;        // causal mask
      }
    }
    float mx = fmaxf(fmaxf(s[0], s[1]), fmaxf(s[2], s[3]));
#pragma unroll
    for (int msk = 1; msk < 16; msk <<= 1)
      mx = fmaxf(mx, __shfl_xor(mx, msk, 32));             // 16-lane half
    const float mnew = fmaxf(mrow[r], mx);
    alpha[r] = __expf(mrow[r] - mnew);
    mrow[r]  = mnew;
#pragma unroll
    for (int nt = 0; nt < 4; ++nt)
      sc[nt][r] = __expf(s[nt] - mnew);
#pragma unroll
    for (int dt = 0; dt < 4; ++dt) acc[dt][r] *= alpha[r];
  }

  // ---- C-tile layout -> A-fragment layout via per-wave LDS round trip
#pragma unroll
  for (int r = 0; r < 8; ++r) {
    const int row = r + hb * 8;
#pragma unroll
    for (int nt = 0; nt < 4; ++nt)
      pw[row * 64 + nt * 16 + l16] = f2bf(sc[nt][r]);
  }
  __builtin_amdgcn_wave_barrier();      // LDS in-order per wave; fence scheduler
  v16bf pa[2];
#pragma unroll
  for (int kk = 0; kk < 2; ++kk) {
    const __bf16* p = pw + l16 * 64 + kk * 32 + hb * 16;
    pa[kk] = *reinterpret_cast<const v16bf*>(p);
  }
  __builtin_amdgcn_wave_barrier();

  // ---- row sums via WMMA: P(16x64) * ones(64x16) -> rowsum in every column
  {
    v8f rs;
#pragma unroll
    for (int e = 0; e < 8; ++e) rs[e] = 0.f;
    rs = wmma_bf16(pa[0], ones, rs);
    rs = wmma_bf16(pa[1], ones, rs);
#pragma unroll
    for (int r = 0; r < 8; ++r)
      lrow[r] = lrow[r] * alpha[r] + rs[r];
  }

  // ---- O += P @ V : V is (B,H,Dk,S) so B-frags are contiguous b128 (8 wmma)
#pragma unroll
  for (int dt = 0; dt < 4; ++dt) {
    const __bf16* vp = vbase + (size_t)(dt * 16 + l16) * SEQ + kb + hb * 16;
    v16bf vb0 = *reinterpret_cast<const v16bf*>(vp);       // keys kb..kb+31
    v16bf vb1 = *reinterpret_cast<const v16bf*>(vp + 32);  // keys kb+32..kb+63
    acc[dt] = wmma_bf16(pa[0], vb0, acc[dt]);
    acc[dt] = wmma_bf16(pa[1], vb1, acc[dt]);
  }
}

// ---------------------------------------------------------------------------
// Causal flash attention. Block = 4 waves = 64 q-rows; each wave owns 16
// q-rows and runs fully independently (per-wave trip count, per-wave LDS
// staging, no block barriers). qrow0 = q0 + 16*wave with q0 a multiple of
// 64  =>  all 64-key blocks with kb < q0 are unmasked and exactly one block
// (kb == q0) needs the causal mask.
// ---------------------------------------------------------------------------
__global__ __launch_bounds__(128)
void attention_kernel(const __bf16* __restrict__ qh, const __bf16* __restrict__ kh,
                      const __bf16* __restrict__ vT, __bf16* __restrict__ ctx) {
  __shared__ __align__(16) __bf16 psh[4][16 * 64];   // per-wave prob staging

  const int lane = threadIdx.x & 31;
  const int wave = threadIdx.x >> 5;
  const int hb   = lane >> 4;
  const int l16  = lane & 15;
  const int bh   = blockIdx.x;          // b*H + h
  const int bb   = bh >> 4;             // NUM_HEADS == 16
  const int hh   = bh & 15;
  const int q0   = blockIdx.y * 64;
  const int qrow0 = q0 + wave * 16;

  const __bf16* qbase = qh + (size_t)bh * SEQ * D_K;
  const __bf16* kbase = kh + (size_t)bh * SEQ * D_K;
  const __bf16* vbase = vT + (size_t)bh * D_K * SEQ;
  __bf16* pw = &psh[wave][0];

  // Q tile 16x64 -> two 16x32 A-fragments (contiguous b128 loads)
  v16bf qa[2];
  {
    const __bf16* qp = qbase + (size_t)(qrow0 + l16) * D_K;
#pragma unroll
    for (int kk = 0; kk < 2; ++kk) {
      const __bf16* p = qp + kk * 32 + hb * 16;
      qa[kk] = *reinterpret_cast<const v16bf*>(p);
    }
  }

  v16bf ones;
  {
    union { v16bf v; unsigned u[8]; } o;
#pragma unroll
    for (int i = 0; i < 8; ++i) o.u[i] = 0x3F803F80u;  // two bf16 1.0s
    ones = o.v;
  }

  v8f   acc[4];
  float mrow[8], lrow[8];
#pragma unroll
  for (int dt = 0; dt < 4; ++dt)
#pragma unroll
    for (int e = 0; e < 8; ++e) acc[dt][e] = 0.f;
#pragma unroll
  for (int r = 0; r < 8; ++r) { mrow[r] = -3.0e38f; lrow[r] = 0.f; }

  // steady state: no masking, no compares
  for (int kb = 0; kb < q0; kb += 64)
    attn_kblock<false>(kb, qrow0, l16, hb, kbase, vbase, pw, qa, ones,
                       acc, mrow, lrow);
  // exactly one partially-masked diagonal block
  attn_kblock<true>(q0, qrow0, l16, hb, kbase, vbase, pw, qa, ones,
                    acc, mrow, lrow);

  // ---- normalize + store ctx (B, S, H*Dk) bf16
#pragma unroll
  for (int r = 0; r < 8; ++r) {
    const float inv = 1.f / lrow[r];
    const int s = qrow0 + r + hb * 8;
    const size_t base = ((size_t)bb * SEQ + s) * D_MODEL + hh * D_K;
#pragma unroll
    for (int dt = 0; dt < 4; ++dt)
      ctx[base + dt * 16 + l16] = f2bf(acc[dt][r] * inv);
  }
}

// ---------------------------------------------------------------------------
// Host-side launch
// ---------------------------------------------------------------------------
extern "C" void kernel_launch(void* const* d_in, const int* in_sizes, int n_in,
                              void* d_out, int out_size, void* d_ws, size_t ws_size,
                              hipStream_t stream) {
  (void)in_sizes; (void)n_in; (void)out_size; (void)ws_size;
  const float* Q  = (const float*)d_in[0];
  const float* Wq = (const float*)d_in[1];
  const float* Wk = (const float*)d_in[2];
  const float* Wv = (const float*)d_in[3];
  const float* Wo = (const float*)d_in[4];
  // d_in[5] is the causal mask; causality is applied analytically.

  char* ws = (char*)d_ws;
  const size_t MB = 1024u * 1024u;
  __bf16* Qb  = (__bf16*)(ws +  0 * MB);   // 16 MB  (8192x1024 bf16)
  __bf16* Wqb = (__bf16*)(ws + 16 * MB);   //  2 MB
  __bf16* Wkb = (__bf16*)(ws + 18 * MB);
  __bf16* Wvb = (__bf16*)(ws + 20 * MB);
  __bf16* Wob = (__bf16*)(ws + 22 * MB);
  __bf16* qh  = (__bf16*)(ws + 24 * MB);   // 16 MB  (B,H,S,Dk)
  __bf16* kh  = (__bf16*)(ws + 40 * MB);   // 16 MB  (B,H,S,Dk)
  __bf16* vT  = (__bf16*)(ws + 56 * MB);   // 16 MB  (B,H,Dk,S)
  __bf16* ctx = (__bf16*)(ws + 72 * MB);   // 16 MB  (B,S,H*Dk)

  const int nQ = BATCH * SEQ * D_MODEL;    // 8388608
  const int nW = D_MODEL * D_MODEL;        // 1048576
  cvt_f32_bf16_kernel<<<nQ / 1024, 256, 0, stream>>>(Q,  Qb,  nQ);
  cvt_f32_bf16_kernel<<<nW / 1024, 256, 0, stream>>>(Wq, Wqb, nW);
  cvt_f32_bf16_kernel<<<nW / 1024, 256, 0, stream>>>(Wk, Wkb, nW);
  cvt_f32_bf16_kernel<<<nW / 1024, 256, 0, stream>>>(Wv, Wvb, nW);
  cvt_f32_bf16_kernel<<<nW / 1024, 256, 0, stream>>>(Wo, Wob, nW);

  dim3 gg(D_MODEL / 128, (BATCH * SEQ) / 128);     // (8, 64)
  gemm_bf16_wmma<0><<<gg, 256, 0, stream>>>(Qb, Wqb, (void*)qh);
  gemm_bf16_wmma<0><<<gg, 256, 0, stream>>>(Qb, Wkb, (void*)kh);
  gemm_bf16_wmma<1><<<gg, 256, 0, stream>>>(Qb, Wvb, (void*)vT);

  dim3 ga(BATCH * NUM_HEADS, SEQ / 64);            // (64, 32)
  attention_kernel<<<ga, 128, 0, stream>>>(qh, kh, vT, ctx);

  gemm_bf16_wmma<2><<<gg, 256, 0, stream>>>(ctx, Wob, d_out);
}
